// JointAttention_31421980738007
// MI455X (gfx1250) — compile-verified, matrix operand order checked by reading, not run on hardware
//
#include <hip/hip_runtime.h>

typedef __attribute__((ext_vector_type(16))) _Float16 v16h;
typedef __attribute__((ext_vector_type(8)))  float    v8f;

#define NS    2
#define BB    2
#define LTOT  2048
#define DM    1024
#define ENC   128
#define NH    16
#define HD    64
#define SJ    1152
#define LENS  1024
#define QSCALE 0.125f   // hd^-0.5

union Frag {
  v16h  v;
  uint4 q[2];
};

static __device__ __forceinline__ v8f wmma16x16x32(const Frag& a, const Frag& b, v8f c) {
  return __builtin_amdgcn_wmma_f32_16x16x32_f16(false, a.v, false, b.v, (short)0, c, false, false);
}

// ---------------------------------------------------------------- f32 -> f16
__global__ void cvt_f16_kernel(const float* __restrict__ src, _Float16* __restrict__ dst, int n) {
  int i = blockIdx.x * blockDim.x + threadIdx.x;
  if (i < n) dst[i] = (_Float16)src[i];
}

// ------------------------------------------------ C[M,N] = A[M,K]*W[N,K]^T + bias
// Register-blocked: each wave computes a 32x64 output block (2 M-tiles x 4 N-tiles),
// 8 f32 accumulators, 12 b128 loads per 8 WMMAs (1.5 loads/WMMA).
__global__ void gemm_f16_kernel(const _Float16* __restrict__ A, const _Float16* __restrict__ W,
                                const float* __restrict__ bias, float* __restrict__ C,
                                int M, int N, int K) {
  int wave   = blockIdx.x * (blockDim.x >> 5) + (threadIdx.x >> 5);
  int tilesN = N >> 6;                      // 64-wide super-tiles
  int tM = wave / tilesN;
  int tN = wave - tM * tilesN;
  if (tM * 32 >= M) return;
  int lane = threadIdx.x & 31;
  int r  = lane & 15;   // row within tile (A) / output col within tile (B)
  int hi = lane >> 4;   // lane-half selects K sub-chunks

  const _Float16* arow0 = A + (size_t)(tM * 32 + r) * K;
  const _Float16* arow1 = arow0 + (size_t)16 * K;
  const _Float16* wrow  = W + (size_t)(tN * 64 + r) * K;

  v8f acc0 = {}, acc1 = {}, acc2 = {}, acc3 = {};
  v8f acc4 = {}, acc5 = {}, acc6 = {}, acc7 = {};

  for (int k0 = 0; k0 < K; k0 += 32) {
    Frag a0, a1, b0, b1, b2, b3;
    a0.q[0] = *reinterpret_cast<const uint4*>(arow0 + k0 +      hi * 8);
    a0.q[1] = *reinterpret_cast<const uint4*>(arow0 + k0 + 16 + hi * 8);
    a1.q[0] = *reinterpret_cast<const uint4*>(arow1 + k0 +      hi * 8);
    a1.q[1] = *reinterpret_cast<const uint4*>(arow1 + k0 + 16 + hi * 8);
    const _Float16* wp = wrow + k0;
    b0.q[0] = *reinterpret_cast<const uint4*>(wp +      hi * 8);
    b0.q[1] = *reinterpret_cast<const uint4*>(wp + 16 + hi * 8);
    wp += (size_t)16 * K;
    b1.q[0] = *reinterpret_cast<const uint4*>(wp +      hi * 8);
    b1.q[1] = *reinterpret_cast<const uint4*>(wp + 16 + hi * 8);
    wp += (size_t)16 * K;
    b2.q[0] = *reinterpret_cast<const uint4*>(wp +      hi * 8);
    b2.q[1] = *reinterpret_cast<const uint4*>(wp + 16 + hi * 8);
    wp += (size_t)16 * K;
    b3.q[0] = *reinterpret_cast<const uint4*>(wp +      hi * 8);
    b3.q[1] = *reinterpret_cast<const uint4*>(wp + 16 + hi * 8);

    acc0 = wmma16x16x32(a0, b0, acc0);
    acc1 = wmma16x16x32(a0, b1, acc1);
    acc2 = wmma16x16x32(a0, b2, acc2);
    acc3 = wmma16x16x32(a0, b3, acc3);
    acc4 = wmma16x16x32(a1, b0, acc4);
    acc5 = wmma16x16x32(a1, b1, acc5);
    acc6 = wmma16x16x32(a1, b2, acc6);
    acc7 = wmma16x16x32(a1, b3, acc7);
  }

  // Straight-line epilogue (no indexed array -> no scratch spill).
  int rowb = tM * 32 + 8 * hi;
  auto store_pair = [&](const v8f& lo, const v8f& hi8, int n) {
    int col = tN * 64 + n * 16 + r;
    float bval = bias[col];
#pragma unroll
    for (int j = 0; j < 8; ++j) {
      int row = rowb + j;                        // C/D layout: M = j + 8*(lane/16)
      C[(size_t)row * N + col]        = lo[j]  + bval;
      C[(size_t)(row + 16) * N + col] = hi8[j] + bval;
    }
  };
  store_pair(acc0, acc4, 0);
  store_pair(acc1, acc5, 1);
  store_pair(acc2, acc6, 2);
  store_pair(acc3, acc7, 3);
}

// -------------------- RMS-norm + RoPE + concat into joint-sequence f16 buffers
// One block (256 thr) per (stage, batch, joint-pos). Thread t owns elements 4t..4t+3.
__global__ void pack_qkv_kernel(const float* __restrict__ Qh, const float* __restrict__ Kh,
                                const float* __restrict__ Vh,
                                const float* __restrict__ Qe, const float* __restrict__ Ke,
                                const float* __restrict__ Ve,
                                const float* __restrict__ gq, const float* __restrict__ gk,
                                const float* __restrict__ gaq, const float* __restrict__ gak,
                                const float* __restrict__ rope,
                                _Float16* __restrict__ cq, _Float16* __restrict__ ck,
                                _Float16* __restrict__ cvT) {
  int idx = blockIdx.x;
  int s  = idx % SJ; idx /= SJ;
  int b  = idx % BB;
  int ip = idx / BB;

  const float *qp, *kp, *vp, *Gq, *Gk;
  if (s < ENC) {
    size_t row = (size_t)((b * NS + ip) * ENC + s) * DM;
    qp = Qe + row; kp = Ke + row; vp = Ve + row; Gq = gaq; Gk = gak;
  } else {
    size_t row = (size_t)(b * LTOT + ip * LENS + (s - ENC)) * DM;
    qp = Qh + row; kp = Kh + row; vp = Vh + row; Gq = gq; Gk = gk;
  }

  int t  = threadIdx.x;
  int e0 = 4 * t;
  float q[4], k[4], v[4];
#pragma unroll
  for (int i = 0; i < 4; ++i) { q[i] = qp[e0 + i]; k[i] = kp[e0 + i]; v[i] = vp[e0 + i]; }

  float sq = 0.f, sk = 0.f;
#pragma unroll
  for (int i = 0; i < 4; ++i) { sq += q[i] * q[i]; sk += k[i] * k[i]; }
#pragma unroll
  for (int m = 1; m < 16; m <<= 1) { sq += __shfl_xor(sq, m, 16); sk += __shfl_xor(sk, m, 16); }
  float rq = rsqrtf(sq * (1.f / 64.f) + 1e-5f);
  float rk = rsqrtf(sk * (1.f / 64.f) + 1e-5f);
#pragma unroll
  for (int i = 0; i < 4; ++i) {
    int d = (e0 + i) & 63;
    q[i] = q[i] * rq * Gq[d];
    k[i] = k[i] * rk * Gk[d];
  }

  // RoPE: f[ip][s][0][i2][j][c], i2 = (elem%64)/2
  const float* f = rope + (size_t)(ip * SJ + s) * 128;
  float qo[4], ko[4];
#pragma unroll
  for (int p = 0; p < 2; ++p) {
    int i2 = ((e0 + 2 * p) & 63) >> 1;
    const float* fp = f + i2 * 4;
    float f00 = fp[0], f01 = fp[1], f10 = fp[2], f11 = fp[3];
    qo[2 * p]     = f00 * q[2 * p] + f01 * q[2 * p + 1];
    qo[2 * p + 1] = f10 * q[2 * p] + f11 * q[2 * p + 1];
    ko[2 * p]     = f00 * k[2 * p] + f01 * k[2 * p + 1];
    ko[2 * p + 1] = f10 * k[2 * p] + f11 * k[2 * p + 1];
  }

  size_t crow = (size_t)((ip * BB + b) * SJ + s) * DM;
#pragma unroll
  for (int i = 0; i < 4; ++i) {
    cq[crow + e0 + i] = (_Float16)qo[i];
    ck[crow + e0 + i] = (_Float16)ko[i];
  }
#pragma unroll
  for (int i = 0; i < 4; ++i) {    // cvT: [ip][b][h][hd][S] (transposed for PV B-operand)
    int h = (e0 + i) >> 6, d = (e0 + i) & 63;
    cvT[(size_t)(((ip * BB + b) * NH + h) * HD + d) * SJ + s] = (_Float16)v[i];
  }
}

// ------------------------------------- flash attention: one 16-row Q tile per wave
__global__ void attn_kernel(const _Float16* __restrict__ cq, const _Float16* __restrict__ ck,
                            const _Float16* __restrict__ cvT, const float* __restrict__ mask,
                            _Float16* __restrict__ Oh, _Float16* __restrict__ Oe) {
  __shared__ __align__(16) _Float16 plds[8 * 16 * 32];   // per-wave 16x32 P staging
  int wave = threadIdx.x >> 5;
  int lane = threadIdx.x & 31;
  int r  = lane & 15;
  int hi = lane >> 4;

  int idx = blockIdx.x;
  int qb = idx % (SJ / 128); idx /= (SJ / 128);
  int h  = idx % NH;         idx /= NH;
  int b  = idx % BB;
  int ip = idx / BB;

  int q0 = (qb * 8 + wave) * 16;
  const _Float16* Qbase = cq  + (size_t)((ip * BB + b) * SJ) * DM + h * HD;
  const _Float16* Kbase = ck  + (size_t)((ip * BB + b) * SJ) * DM + h * HD;
  const _Float16* Vbase = cvT + (size_t)(((ip * BB + b) * NH + h) * HD) * SJ;
  const float*    Mbase = mask + (size_t)(ip * BB + b) * SJ * SJ;

  Frag aq0, aq1;   // Q A-fragments for hd = 0..31 and 32..63
  {
    const _Float16* qrow = Qbase + (size_t)(q0 + r) * DM;
    aq0.q[0] = *reinterpret_cast<const uint4*>(qrow +      hi * 8);
    aq0.q[1] = *reinterpret_cast<const uint4*>(qrow + 16 + hi * 8);
    aq1.q[0] = *reinterpret_cast<const uint4*>(qrow + 32 + hi * 8);
    aq1.q[1] = *reinterpret_cast<const uint4*>(qrow + 48 + hi * 8);
  }

  float mrow[8], lrow[8];
  v8f o0 = {}, o1 = {}, o2 = {}, o3 = {};
#pragma unroll
  for (int j = 0; j < 8; ++j) { mrow[j] = -3.0e38f; lrow[j] = 0.f; }

  _Float16* P = plds + wave * (16 * 32);

  for (int kb = 0; kb < SJ; kb += 32) {
    v8f s0 = {}, s1 = {};
    {
      const _Float16* krow = Kbase + (size_t)(kb + r) * DM;
      Frag fb;
      fb.q[0] = *reinterpret_cast<const uint4*>(krow +      hi * 8);
      fb.q[1] = *reinterpret_cast<const uint4*>(krow + 16 + hi * 8);
      s0 = wmma16x16x32(aq0, fb, s0);
      fb.q[0] = *reinterpret_cast<const uint4*>(krow + 32 + hi * 8);
      fb.q[1] = *reinterpret_cast<const uint4*>(krow + 48 + hi * 8);
      s0 = wmma16x16x32(aq1, fb, s0);
      krow += (size_t)16 * DM;
      fb.q[0] = *reinterpret_cast<const uint4*>(krow +      hi * 8);
      fb.q[1] = *reinterpret_cast<const uint4*>(krow + 16 + hi * 8);
      s1 = wmma16x16x32(aq0, fb, s1);
      fb.q[0] = *reinterpret_cast<const uint4*>(krow + 32 + hi * 8);
      fb.q[1] = *reinterpret_cast<const uint4*>(krow + 48 + hi * 8);
      s1 = wmma16x16x32(aq1, fb, s1);
    }
    // online softmax over this 32-key block (row lives across 16 lanes, same hi)
#pragma unroll
    for (int j = 0; j < 8; ++j) {
      int qq = q0 + j + 8 * hi;
      const float* mp = Mbase + (size_t)qq * SJ + kb;
      float v0 = s0[j] * QSCALE + mp[r];
      float v1 = s1[j] * QSCALE + mp[16 + r];
      float tmax = fmaxf(v0, v1);
#pragma unroll
      for (int m = 1; m < 16; m <<= 1) tmax = fmaxf(tmax, __shfl_xor(tmax, m, 16));
      float mnew = fmaxf(mrow[j], tmax);
      float corr = __expf(mrow[j] - mnew);
      float p0 = __expf(v0 - mnew);
      float p1 = __expf(v1 - mnew);
      float ps = p0 + p1;
#pragma unroll
      for (int m = 1; m < 16; m <<= 1) ps += __shfl_xor(ps, m, 16);
      lrow[j] = lrow[j] * corr + ps;
      mrow[j] = mnew;
      o0[j] *= corr; o1[j] *= corr; o2[j] *= corr; o3[j] *= corr;
      int pr = (j + 8 * hi) * 32;                   // C-layout -> LDS row-major 16x32
      P[pr + r]      = (_Float16)p0;
      P[pr + 16 + r] = (_Float16)p1;
    }
    Frag ap;   // re-read P in A-fragment layout (LDS transpose)
    {
      const _Float16* prow = P + r * 32;
      ap.q[0] = *reinterpret_cast<const uint4*>(prow +      hi * 8);
      ap.q[1] = *reinterpret_cast<const uint4*>(prow + 16 + hi * 8);
    }
    {
      const _Float16* vrow = Vbase + (size_t)r * SJ + kb;
      Frag fb;
      fb.q[0] = *reinterpret_cast<const uint4*>(vrow +      hi * 8);
      fb.q[1] = *reinterpret_cast<const uint4*>(vrow + 16 + hi * 8);
      o0 = wmma16x16x32(ap, fb, o0);
      vrow += (size_t)16 * SJ;
      fb.q[0] = *reinterpret_cast<const uint4*>(vrow +      hi * 8);
      fb.q[1] = *reinterpret_cast<const uint4*>(vrow + 16 + hi * 8);
      o1 = wmma16x16x32(ap, fb, o1);
      vrow += (size_t)16 * SJ;
      fb.q[0] = *reinterpret_cast<const uint4*>(vrow +      hi * 8);
      fb.q[1] = *reinterpret_cast<const uint4*>(vrow + 16 + hi * 8);
      o2 = wmma16x16x32(ap, fb, o2);
      vrow += (size_t)16 * SJ;
      fb.q[0] = *reinterpret_cast<const uint4*>(vrow +      hi * 8);
      fb.q[1] = *reinterpret_cast<const uint4*>(vrow + 16 + hi * 8);
      o3 = wmma16x16x32(ap, fb, o3);
    }
  }

#pragma unroll
  for (int j = 0; j < 8; ++j) {
    float inv = 1.f / lrow[j];
    int s = q0 + j + 8 * hi;
    _Float16* dst;
    if (s < ENC) dst = Oe + (size_t)((b * NS + ip) * ENC + s) * DM + h * HD + r;
    else         dst = Oh + (size_t)(b * LTOT + ip * LENS + (s - ENC)) * DM + h * HD + r;
    dst[0]  = (_Float16)(o0[j] * inv);
    dst[16] = (_Float16)(o1[j] * inv);
    dst[32] = (_Float16)(o2[j] * inv);
    dst[48] = (_Float16)(o3[j] * inv);
  }
}

// ---------------------------------------------------------------------- host
extern "C" void kernel_launch(void* const* d_in, const int* in_sizes, int n_in,
                              void* d_out, int out_size, void* d_ws, size_t ws_size,
                              hipStream_t stream) {
  (void)in_sizes; (void)n_in; (void)out_size; (void)ws_size;
  const float* Xh  = (const float*)d_in[0];
  const float* Xe  = (const float*)d_in[1];
  const float* Msk = (const float*)d_in[2];
  const float* Rp  = (const float*)d_in[3];
  const float* Wq  = (const float*)d_in[4];  const float* bq  = (const float*)d_in[5];
  const float* Wk  = (const float*)d_in[6];  const float* bk  = (const float*)d_in[7];
  const float* Wv  = (const float*)d_in[8];  const float* bv  = (const float*)d_in[9];
  const float* Waq = (const float*)d_in[10]; const float* baq = (const float*)d_in[11];
  const float* Wak = (const float*)d_in[12]; const float* bak = (const float*)d_in[13];
  const float* Wav = (const float*)d_in[14]; const float* bav = (const float*)d_in[15];
  const float* Wo  = (const float*)d_in[16]; const float* bo  = (const float*)d_in[17];
  const float* Wao = (const float*)d_in[18]; const float* bao = (const float*)d_in[19];
  const float* gq  = (const float*)d_in[20]; const float* gk  = (const float*)d_in[21];
  const float* gaq = (const float*)d_in[22]; const float* gak = (const float*)d_in[23];

  const int ROWS_H = BB * LTOT;           // 4096
  const int ROWS_E = BB * NS * ENC;       // 512
  const int NW     = DM * DM;             // 1048576
  const size_t CN  = (size_t)NS * BB * SJ * DM;

  char* ws = (char*)d_ws;
  size_t off = 0;
  auto take = [&](size_t bytes) -> void* {
    void* p = (void*)(ws + off);
    off += (bytes + 255) & ~(size_t)255;
    return p;
  };

  _Float16* hWq  = (_Float16*)take((size_t)NW * 2);
  _Float16* hWk  = (_Float16*)take((size_t)NW * 2);
  _Float16* hWv  = (_Float16*)take((size_t)NW * 2);
  _Float16* hWaq = (_Float16*)take((size_t)NW * 2);
  _Float16* hWak = (_Float16*)take((size_t)NW * 2);
  _Float16* hWav = (_Float16*)take((size_t)NW * 2);
  _Float16* hWo  = (_Float16*)take((size_t)NW * 2);
  _Float16* hWao = (_Float16*)take((size_t)NW * 2);
  _Float16* hXh  = (_Float16*)take((size_t)ROWS_H * DM * 2);
  _Float16* hXe  = (_Float16*)take((size_t)ROWS_E * DM * 2);
  float* Qh = (float*)take((size_t)ROWS_H * DM * 4);
  float* Kh = (float*)take((size_t)ROWS_H * DM * 4);
  float* Vh = (float*)take((size_t)ROWS_H * DM * 4);
  float* Qe = (float*)take((size_t)ROWS_E * DM * 4);
  float* Ke = (float*)take((size_t)ROWS_E * DM * 4);
  float* Ve = (float*)take((size_t)ROWS_E * DM * 4);
  _Float16* cq  = (_Float16*)take(CN * 2);
  _Float16* ck  = (_Float16*)take(CN * 2);
  _Float16* cvT = (_Float16*)take(CN * 2);
  _Float16* Oh  = (_Float16*)take((size_t)ROWS_H * DM * 2);
  _Float16* Oe  = (_Float16*)take((size_t)ROWS_E * DM * 2);

  auto cvt = [&](const float* s, _Float16* d, int n) {
    cvt_f16_kernel<<<dim3((n + 255) / 256), dim3(256), 0, stream>>>(s, d, n);
  };
  auto gemm = [&](const _Float16* A, const _Float16* W, const float* bias, float* C,
                  int M, int N, int K) {
    int waves = (M / 32) * (N / 64);      // one 32x64 block per wave
    gemm_f16_kernel<<<dim3((waves + 7) / 8), dim3(256), 0, stream>>>(A, W, bias, C, M, N, K);
  };

  cvt(Wq, hWq, NW);   cvt(Wk, hWk, NW);   cvt(Wv, hWv, NW);
  cvt(Waq, hWaq, NW); cvt(Wak, hWak, NW); cvt(Wav, hWav, NW);
  cvt(Wo, hWo, NW);   cvt(Wao, hWao, NW);
  cvt(Xh, hXh, ROWS_H * DM);
  cvt(Xe, hXe, ROWS_E * DM);

  gemm(hXh, hWq, bq, Qh, ROWS_H, DM, DM);
  gemm(hXh, hWk, bk, Kh, ROWS_H, DM, DM);
  gemm(hXh, hWv, bv, Vh, ROWS_H, DM, DM);
  gemm(hXe, hWaq, baq, Qe, ROWS_E, DM, DM);
  gemm(hXe, hWak, bak, Ke, ROWS_E, DM, DM);
  gemm(hXe, hWav, bav, Ve, ROWS_E, DM, DM);

  pack_qkv_kernel<<<dim3(NS * BB * SJ), dim3(256), 0, stream>>>(
      Qh, Kh, Vh, Qe, Ke, Ve, gq, gk, gaq, gak, Rp, cq, ck, cvT);

  attn_kernel<<<dim3(NS * BB * NH * (SJ / 128)), dim3(256), 0, stream>>>(
      cq, ck, cvT, Msk, Oh, Oe);

  float* out = (float*)d_out;
  gemm(Oh, hWo, bo, out, ROWS_H, DM, DM);
  gemm(Oe, hWao, bao, out + (size_t)ROWS_H * DM, ROWS_E, DM, DM);
}
